// Encoder_14396730376941
// MI455X (gfx1250) — matrix operand order repaired and checked by reading, not compiled, not run on hardware
//
#include <hip/hip_runtime.h>
#include <hip/hip_bf16.h>
#include <math.h>

// ---------------------------------------------------------------------------
// MI455X (gfx1250) bidirectional-GRU encoder.
//  * All GEMMs on v_wmma_f32_16x16x32_bf16 (fp32 accumulate, bf16 operands).
//  * GEMM blocks cover 64 M-rows: activations staged once in LDS (bf16,
//    padded stride -> conflict-free ds_load_b128 A-fragments), weight
//    B-fragments held in VGPRs across all 4 M-tiles (4x less L2 traffic).
//  * Recurrence: persistent block per (dir, 16-batch-row tile); h state in
//    LDS; whh WMMA B-fragments live in registers for all 200 steps.
//  * Output projections fused with sqrt(exp(.)) epilogues into d_out.
// ---------------------------------------------------------------------------

typedef __attribute__((ext_vector_type(16))) __bf16 v16bf;
typedef __attribute__((ext_vector_type(8)))  float v8f;

#define T_LEN 200
#define BATCH 1024
#define D_IN  256
#define H     128
#define G3    384                                   // 3*H
#define XP_DIR_STRIDE ((size_t)BATCH * T_LEN * G3)  // 78,643,200 elems
#define LDSTRIDE 264   // bf16 elems per staged row (256 + 8 pad: bank rotate)

// ----- bf16 helpers (round-to-nearest-even) --------------------------------
__device__ __forceinline__ unsigned int bf_round(unsigned int u) {
  return (u + 0x7FFFu + ((u >> 16) & 1u)) >> 16;
}
__device__ __forceinline__ unsigned short f2bf(float f) {
  union { float f; unsigned u; } x; x.f = f;
  return (unsigned short)bf_round(x.u);
}
__device__ __forceinline__ unsigned int pk2(float lo, float hi) {
  union { float f; unsigned u; } a, b; a.f = lo; b.f = hi;
  return (bf_round(a.u) & 0xFFFFu) | (bf_round(b.u) << 16);
}
__device__ __forceinline__ float bf2f(unsigned short h) {
  union { unsigned u; float f; } x; x.u = ((unsigned)h) << 16; return x.f;
}

// ----- WMMA fragment (16x16x32 bf16): 8 dwords / lane ----------------------
union Frag {
  v16bf    v;
  unsigned u[8];
};

__device__ __forceinline__ v8f wmma_bf16(const Frag& a, const Frag& b, v8f c) {
  return __builtin_amdgcn_wmma_f32_16x16x32_bf16(
      /*neg_a=*/false, a.v, /*neg_b=*/false, b.v,
      /*c_mod=*/(short)0, c, /*reuse_a=*/false, /*reuse_b=*/false);
}

// A fragment (16xK tile, row M = lane&15) from fp32 row pointer.
// ISA layout: lanes<16 -> K base 0, lanes>=16 -> K base 8; pairs at
// base+{0..7} and base+16+{0..7} within the 32-wide K tile.
__device__ __forceinline__ void loadA_f32(const float* row, int kt, int lane, Frag& f) {
  const float* p = row + kt * 32 + ((lane & 16) ? 8 : 0);
  float4 x0 = *(const float4*)(p);
  float4 x1 = *(const float4*)(p + 4);
  float4 x2 = *(const float4*)(p + 16);
  float4 x3 = *(const float4*)(p + 20);
  f.u[0] = pk2(x0.x, x0.y); f.u[1] = pk2(x0.z, x0.w);
  f.u[2] = pk2(x1.x, x1.y); f.u[3] = pk2(x1.z, x1.w);
  f.u[4] = pk2(x2.x, x2.y); f.u[5] = pk2(x2.z, x2.w);
  f.u[6] = pk2(x3.x, x3.y); f.u[7] = pk2(x3.z, x3.w);
}

// A fragment from bf16 row pointer (global or LDS); tile at elem offset kbase.
__device__ __forceinline__ void loadA_bf16(const unsigned short* row, int kbase,
                                           int lane, Frag& f) {
  const unsigned* p = (const unsigned*)(row) + (kbase >> 1) + ((lane & 16) ? 4 : 0);
  uint4 q0 = *(const uint4*)(p);
  uint4 q1 = *(const uint4*)(p + 8);
  f.u[0] = q0.x; f.u[1] = q0.y; f.u[2] = q0.z; f.u[3] = q0.w;
  f.u[4] = q1.x; f.u[5] = q1.y; f.u[6] = q1.z; f.u[7] = q1.w;
}

// B fragment (Kx16, column N = lane&15) from bf16 weight matrix, row g
// (row-major [g][k], ldk = K stride). lanes<16 -> K base 0, lanes>=16 ->
// K base 8; base+{0..7} and base+16+{0..7} of the 32-wide K tile.
__device__ __forceinline__ void loadB(const unsigned short* wmat, int ldk, int g,
                                      int kt, int lane, Frag& f) {
  int kb = kt * 32 + ((lane & 16) ? 8 : 0);
  const uint4* p = (const uint4*)(wmat + (size_t)g * ldk + kb);
  uint4 q0 = p[0], q1 = p[2];
  f.u[0] = q0.x; f.u[1] = q0.y; f.u[2] = q0.z; f.u[3] = q0.w;
  f.u[4] = q1.x; f.u[5] = q1.y; f.u[6] = q1.z; f.u[7] = q1.w;
}

// ===========================================================================
// Kernel: fp32 -> bf16 weight conversion
// ===========================================================================
__global__ void k_f32_to_bf16(const float* __restrict__ src,
                              unsigned short* __restrict__ dst, int n) {
  int i = blockIdx.x * blockDim.x + threadIdx.x;
  if (i < n) dst[i] = f2bf(src[i]);
}

// ===========================================================================
// Kernel: xp[dir] = data @ wih[dir].T + bih[dir]   (M=204800, K=256, N=4x384)
// Block = 64 M-rows staged in LDS (bf16).  8 waves; wave w: dir=w>>1,
// ntiles (w&1)*12..+11.  B fragments register-resident across 4 M-tiles.
// ===========================================================================
__global__ __launch_bounds__(256) void k_xp_gemm(
    const float* __restrict__ data,
    const unsigned short* __restrict__ wih_bf,   // 4 * 384*256 bf16
    const float* __restrict__ bih0, const float* __restrict__ bih1,
    const float* __restrict__ bih2, const float* __restrict__ bih3,
    unsigned short* __restrict__ xp) {
  __shared__ unsigned sh_u[64 * (LDSTRIDE / 2)];   // 64 rows x 256 bf16 (33 KB)

  int tid  = threadIdx.x;
  int lane = tid & 31;
  int wv   = tid >> 5;
  int n15  = lane & 15;
  int hi8  = (lane & 16) ? 8 : 0;
  int row0 = blockIdx.x * 64;

  // ---- stage A tile: 64x256 fp32 -> bf16 LDS (padded stride) ----
  for (int i = tid; i < 64 * 128; i += 256) {
    int r = i >> 7, cd = i & 127;                  // cd = dword col (2 bf16)
    const float2 v = *(const float2*)(data + (size_t)(row0 + r) * D_IN + cd * 2);
    sh_u[r * (LDSTRIDE / 2) + cd] = pk2(v.x, v.y);
  }
  __syncthreads();

  int dir = wv >> 1;                               // uniform per wave
  const float* bih = (dir == 0) ? bih0 : (dir == 1) ? bih1
                    : (dir == 2) ? bih2 : bih3;
  const unsigned short* wm = wih_bf + (size_t)dir * (G3 * D_IN);
  unsigned short* xpd = xp + (size_t)dir * XP_DIR_STRIDE;

#pragma unroll 1
  for (int p = 0; p < 12; ++p) {
    int nt = (wv & 1) * 12 + p;
    int g  = nt * 16 + n15;
    Frag Bf[8];                                    // weights pinned in VGPRs
#pragma unroll
    for (int kt = 0; kt < 8; ++kt) loadB(wm, D_IN, g, kt, lane, Bf[kt]);
    float bv = bih[g];
#pragma unroll 1
    for (int mt = 0; mt < 4; ++mt) {               // reuse B over 4 M-tiles
      const unsigned short* arow =
          (const unsigned short*)sh_u + (mt * 16 + n15) * LDSTRIDE;
      v8f acc = {};
#pragma unroll
      for (int kt = 0; kt < 8; ++kt) {
        Frag Af; loadA_bf16(arow, kt * 32, lane, Af);
        acc = wmma_bf16(Af, Bf[kt], acc);
      }
#pragma unroll
      for (int i = 0; i < 8; ++i) {
        size_t m = (size_t)row0 + mt * 16 + i + hi8;  // C/D: VGPR i -> M=i/i+8
        xpd[m * G3 + g] = f2bf(acc[i] + bv);
      }
    }
  }
}

// ===========================================================================
// Kernel: persistent GRU recurrence.  grid = 4 dirs * 64 batch tiles.
// Per step: hp = h @ whh.T + bhh via WMMA (12 wmma/wave), then fp32 gate
// math through LDS.  whh B-fragments live in registers for all 200 steps.
// ===========================================================================
__global__ __launch_bounds__(256) void k_gru(
    const unsigned short* __restrict__ xp,
    const unsigned short* __restrict__ whh_bf,   // 4 * 384*128 bf16
    const float* __restrict__ bhh0, const float* __restrict__ bhh1,
    const float* __restrict__ bhh2, const float* __restrict__ bhh3,
    const float* __restrict__ h0_ge, const float* __restrict__ h0_ne,
    unsigned short* __restrict__ hs_f, unsigned short* __restrict__ hs_b,
    float* __restrict__ hlast) {
  __shared__ float sh_h[16 * H];     // fp32 hidden state (8 KB)
  __shared__ float sh_hp[16 * G3];   // fp32 hp = h@whh.T + bhh (24 KB)

  int tid  = threadIdx.x;
  int lane = tid & 31;
  int wv   = tid >> 5;
  int n15  = lane & 15;
  int hi8  = (lane & 16) ? 8 : 0;

  int  dir   = blockIdx.x >> 6;       // 0=ge_f 1=ge_b 2=ne_f 3=ne_b
  int  mtile = blockIdx.x & 63;
  int  b0    = mtile * 16;
  bool rev   = (dir & 1) != 0;
  bool isNe  = dir >= 2;

  const unsigned short* xpd = xp + (size_t)dir * XP_DIR_STRIDE;
  const float* bhh = (dir == 0) ? bhh0 : (dir == 1) ? bhh1
                    : (dir == 2) ? bhh2 : bhh3;
  const float* h0 = (isNe ? h0_ne : h0_ge) + (size_t)(dir & 1) * H;
  unsigned short* hs = isNe ? ((dir == 2) ? hs_f : hs_b) : (unsigned short*)0;

  // init hidden state (broadcast h0 row)
  for (int p = tid; p < 16 * H; p += 256) sh_h[p] = h0[p & (H - 1)];

  // preload whh B-fragments: 3 N-tiles x 4 K-tiles, held in VGPRs 200 steps
  const unsigned short* wm = whh_bf + (size_t)dir * (G3 * H);
  Frag Bw[3][4];
#pragma unroll
  for (int p = 0; p < 3; ++p) {
    int g = (wv * 3 + p) * 16 + n15;
#pragma unroll
    for (int kt = 0; kt < 4; ++kt) loadB(wm, H, g, kt, lane, Bw[p][kt]);
  }
  __syncthreads();

  for (int s = 0; s < T_LEN; ++s) {
    int tt = rev ? (T_LEN - 1 - s) : s;

    // ---- phase 1: hp = h @ whh.T + bhh  (WMMA) ----
    Frag A[4];
    const float* hrow = &sh_h[n15 * H];
#pragma unroll
    for (int kt = 0; kt < 4; ++kt) loadA_f32(hrow, kt, lane, A[kt]);
#pragma unroll
    for (int p = 0; p < 3; ++p) {
      int nt = wv * 3 + p;
      int g  = nt * 16 + n15;
      v8f acc = {};
#pragma unroll
      for (int kt = 0; kt < 4; ++kt) acc = wmma_bf16(A[kt], Bw[p][kt], acc);
      float bv = bhh[g];
#pragma unroll
      for (int i = 0; i < 8; ++i) sh_hp[(i + hi8) * G3 + g] = acc[i] + bv;
    }
    __syncthreads();

    // ---- phase 2: gate math (fp32) ----
#pragma unroll 1
    for (int p = 0; p < 8; ++p) {
      int idx = tid + p * 256;           // 16*128 = 2048 (m, j) items
      int m = idx >> 7, j = idx & 127;
      size_t xoff = ((size_t)(b0 + m) * T_LEN + tt) * G3;
      float xr = bf2f(xpd[xoff + j]);
      float xz = bf2f(xpd[xoff + H + j]);
      float xn = bf2f(xpd[xoff + 2 * H + j]);
      float hr = sh_hp[m * G3 + j];
      float hz = sh_hp[m * G3 + H + j];
      float hn = sh_hp[m * G3 + 2 * H + j];
      float r  = 1.0f / (1.0f + __expf(-(xr + hr)));
      float z  = 1.0f / (1.0f + __expf(-(xz + hz)));
      float nn = tanhf(xn + r * hn);
      float hprev = sh_h[m * H + j];
      float hnew  = (1.0f - z) * nn + z * hprev;
      hnew = fminf(5.0f, fmaxf(-5.0f, hnew));
      sh_h[m * H + j] = hnew;
      if (isNe) hs[((size_t)(b0 + m) * T_LEN + tt) * H + j] = f2bf(hnew);
    }
    // warm L2/WGP$ for next step's xp rows (global_prefetch_b8)
    if (s + 1 < T_LEN) {
      int tn = rev ? (tt - 1) : (tt + 1);
      int pidx = tid * 8;
      int pm = pidx >> 7, pj = pidx & 127;
      __builtin_prefetch(xpd + ((size_t)(b0 + pm) * T_LEN + tn) * G3 + pj, 0, 1);
    }
    __syncthreads();
  }

  if (!isNe) {  // ge dirs: only the final hidden state is consumed
    size_t half = (size_t)(dir & 1) * BATCH * H;
    for (int p = tid; p < 16 * H; p += 256) {
      int m = p >> 7, j = p & 127;
      hlast[half + (size_t)(b0 + m) * H + j] = sh_h[p];
    }
  }
}

// ===========================================================================
// Kernel: input_params = [hs_f | hs_b] @ inp_w.T + inp_b, fused epilogue
// mean -> d_out[262144 ..), std = sqrt(exp(logvar)) -> next region.
// Block = 64 M-rows staged in LDS; B fragments reused over 4 M-tiles.
// ===========================================================================
__global__ __launch_bounds__(256) void k_hs_proj(
    const unsigned short* __restrict__ hs_f,
    const unsigned short* __restrict__ hs_b,
    const unsigned short* __restrict__ inpw_bf,  // 256*256 bf16
    const float* __restrict__ inp_b,
    float* __restrict__ out) {
  __shared__ unsigned sh_u[64 * (LDSTRIDE / 2)];  // 64 rows x 256 bf16 (33 KB)

  int tid  = threadIdx.x;
  int lane = tid & 31;
  int wv   = tid >> 5;
  int n15  = lane & 15;
  int hi8  = (lane & 16) ? 8 : 0;
  int row0 = blockIdx.x * 64;

  // ---- stage A tile: [hs_f | hs_b] concat along K (already bf16) ----
  const unsigned* uf = (const unsigned*)hs_f;
  const unsigned* ub = (const unsigned*)hs_b;
  for (int i = tid; i < 64 * 128; i += 256) {
    int r = i >> 7, cd = i & 127;
    unsigned v = (cd < 64) ? uf[(size_t)(row0 + r) * 64 + cd]
                           : ub[(size_t)(row0 + r) * 64 + (cd - 64)];
    sh_u[r * (LDSTRIDE / 2) + cd] = v;
  }
  __syncthreads();

  float* omean = out + 262144;
  float* ostd  = out + 262144 + (size_t)BATCH * T_LEN * H;

#pragma unroll 1
  for (int p = 0; p < 2; ++p) {
    int nt = wv * 2 + p;
    int c  = nt * 16 + n15;
    Frag Bf[8];
#pragma unroll
    for (int kt = 0; kt < 8; ++kt) loadB(inpw_bf, 2 * H, c, kt, lane, Bf[kt]);
    float bv = inp_b[c];
#pragma unroll 1
    for (int mt = 0; mt < 4; ++mt) {
      const unsigned short* arow =
          (const unsigned short*)sh_u + (mt * 16 + n15) * LDSTRIDE;
      v8f acc = {};
#pragma unroll
      for (int kt = 0; kt < 8; ++kt) {
        Frag Af; loadA_bf16(arow, kt * 32, lane, Af);
        acc = wmma_bf16(Af, Bf[kt], acc);
      }
#pragma unroll
      for (int i = 0; i < 8; ++i) {
        size_t m = (size_t)row0 + mt * 16 + i + hi8;
        float v = acc[i] + bv;
        if (c < H) omean[m * H + c] = v;
        else       ostd[m * H + (c - H)] = sqrtf(__expf(v));
      }
    }
  }
}

// ===========================================================================
// Kernel: init projection (tiny: 1024 x 256 x 256) — plain fp32.
// ===========================================================================
__global__ __launch_bounds__(256) void k_init_proj(
    const float* __restrict__ hlast,     // [2][1024][128] fp32 (f then b)
    const float* __restrict__ init_w,    // (256, 256) fp32
    const float* __restrict__ init_b,
    float* __restrict__ out) {
  int t = blockIdx.x * blockDim.x + threadIdx.x;   // 1024*256
  int b = t >> 8, c = t & 255;
  const float* hf = hlast + (size_t)b * H;
  const float* hb = hlast + (size_t)BATCH * H + (size_t)b * H;
  const float* wr = init_w + (size_t)c * 256;
  float acc = init_b[c];
#pragma unroll 4
  for (int k = 0; k < H; ++k) acc += hf[k] * wr[k];
#pragma unroll 4
  for (int k = 0; k < H; ++k) acc += hb[k] * wr[H + k];
  if (c < H) out[(size_t)b * H + c] = acc;
  else       out[131072 + (size_t)b * H + (c - H)] = sqrtf(__expf(acc) + 1e-4f);
}

// ===========================================================================
extern "C" void kernel_launch(void* const* d_in, const int* in_sizes, int n_in,
                              void* d_out, int out_size, void* d_ws, size_t ws_size,
                              hipStream_t stream) {
  (void)in_sizes; (void)n_in; (void)out_size; (void)ws_size;

  const float* data  = (const float*)d_in[0];
  const float* h0_ge = (const float*)d_in[1];
  const float* h0_ne = (const float*)d_in[2];
  const float* wih[4]; const float* whh[4]; const float* bih[4]; const float* bhh[4];
  for (int d = 0; d < 4; ++d) {
    wih[d] = (const float*)d_in[3 + 4 * d];
    whh[d] = (const float*)d_in[4 + 4 * d];
    bih[d] = (const float*)d_in[5 + 4 * d];
    bhh[d] = (const float*)d_in[6 + 4 * d];
  }
  const float* init_w = (const float*)d_in[19];
  const float* init_b = (const float*)d_in[20];
  const float* inp_w  = (const float*)d_in[21];
  const float* inp_b  = (const float*)d_in[22];
  float* out = (float*)d_out;

  // ---- workspace carve-up (256B aligned) ----
  char* ws = (char*)d_ws;
  size_t off = 0;
  auto carve = [&](size_t bytes) {
    void* p = ws + off;
    off = (off + bytes + 255) & ~(size_t)255;
    return p;
  };
  unsigned short* wih_bf  = (unsigned short*)carve((size_t)4 * G3 * D_IN * 2);
  unsigned short* whh_bf  = (unsigned short*)carve((size_t)4 * G3 * H * 2);
  unsigned short* inpw_bf = (unsigned short*)carve((size_t)256 * 256 * 2);
  unsigned short* xp      = (unsigned short*)carve((size_t)4 * XP_DIR_STRIDE * 2);
  unsigned short* hsf     = (unsigned short*)carve((size_t)BATCH * T_LEN * H * 2);
  unsigned short* hsb     = (unsigned short*)carve((size_t)BATCH * T_LEN * H * 2);
  float*          hlast   = (float*)carve((size_t)2 * BATCH * H * 4);

  // ---- 1) weight conversion to bf16 ----
  for (int d = 0; d < 4; ++d) {
    k_f32_to_bf16<<<(G3 * D_IN) / 256, 256, 0, stream>>>(
        wih[d], wih_bf + (size_t)d * G3 * D_IN, G3 * D_IN);
    k_f32_to_bf16<<<(G3 * H) / 256, 256, 0, stream>>>(
        whh[d], whh_bf + (size_t)d * G3 * H, G3 * H);
  }
  k_f32_to_bf16<<<(256 * 256) / 256, 256, 0, stream>>>(inp_w, inpw_bf, 256 * 256);

  // ---- 2) xp = data @ wih.T + bih (all 4 dirs), 64-row blocks ----
  k_xp_gemm<<<(BATCH * T_LEN) / 64, 256, 0, stream>>>(
      data, wih_bf, bih[0], bih[1], bih[2], bih[3], xp);

  // ---- 3) persistent GRU recurrence (4 dirs x 64 batch tiles) ----
  k_gru<<<4 * (BATCH / 16), 256, 0, stream>>>(
      xp, whh_bf, bhh[0], bhh[1], bhh[2], bhh[3],
      h0_ge, h0_ne, hsf, hsb, hlast);

  // ---- 4) input projection + epilogue, 64-row blocks ----
  k_hs_proj<<<(BATCH * T_LEN) / 64, 256, 0, stream>>>(hsf, hsb, inpw_bf, inp_b, out);

  // ---- 5) init projection + epilogue ----
  k_init_proj<<<(BATCH * 256) / 256, 256, 0, stream>>>(hlast, init_w, init_b, out);
}
